// CounterPropagationNetwork_6605659701676
// MI455X (gfx1250) — matrix operand order repaired and checked by reading, not compiled, not run on hardware
//
#include <hip/hip_runtime.h>
#include <hip/hip_bf16.h>

#define BATCH 4096
#define IN_   2048
#define HID_  8192
#define NCLS  1000
#define NSPLIT 4                       // N-dimension split for occupancy
#define NT_PER_SPLIT (HID_ / 16 / NSPLIT)   // 128 N-tiles per split

// LDS row stride in halfs: pad by 8 halfs (16B) so 16 lanes reading 16
// different rows at the same K hit different banks.
#define A_STRIDE (IN_ + 8)

typedef __bf16 bf16;
typedef bf16  v16bf __attribute__((ext_vector_type(16)));
typedef float v8f   __attribute__((ext_vector_type(8)));

static __device__ __forceinline__ unsigned short f2bf(float f) {
  union { float f; unsigned u; } a; a.f = f;
  unsigned u = a.u;
  unsigned r = u + 0x7FFFu + ((u >> 16) & 1u);   // round-to-nearest-even
  return (unsigned short)(r >> 16);
}

static __device__ __forceinline__ uint4 pack8(const float* v) {
  unsigned short h[8];
  #pragma unroll
  for (int i = 0; i < 8; ++i) h[i] = f2bf(v[i]);
  uint4 o;
  o.x = (unsigned)h[0] | ((unsigned)h[1] << 16);
  o.y = (unsigned)h[2] | ((unsigned)h[3] << 16);
  o.z = (unsigned)h[4] | ((unsigned)h[5] << 16);
  o.w = (unsigned)h[6] | ((unsigned)h[7] << 16);
  return o;
}

// ---------------------------------------------------------------------------
// Kernel 1: L2-normalize each row of x, emit bf16. One block (8 waves) per row.
// ---------------------------------------------------------------------------
__global__ __launch_bounds__(256) void normalize_cvt(
    const float* __restrict__ x, unsigned short* __restrict__ xn) {
  const int b    = blockIdx.x;
  const int t    = threadIdx.x;
  const int lane = t & 31;
  const int wave = t >> 5;

  const float4* row = (const float4*)(x + (size_t)b * IN_);
  float4 v0 = row[t * 2 + 0];
  float4 v1 = row[t * 2 + 1];
  float f[8] = { v0.x, v0.y, v0.z, v0.w, v1.x, v1.y, v1.z, v1.w };

  float ss = 0.f;
  #pragma unroll
  for (int i = 0; i < 8; ++i) ss += f[i] * f[i];
  #pragma unroll
  for (int off = 16; off >= 1; off >>= 1) ss += __shfl_xor(ss, off, 32);

  __shared__ float sacc[8];
  if (lane == 0) sacc[wave] = ss;
  __syncthreads();
  float tot = 0.f;
  #pragma unroll
  for (int w = 0; w < 8; ++w) tot += sacc[w];
  const float inv = rsqrtf(tot);

  #pragma unroll
  for (int i = 0; i < 8; ++i) f[i] *= inv;
  ((uint4*)(xn + (size_t)b * IN_))[t] = pack8(f);
}

// ---------------------------------------------------------------------------
// Kernel 2: Wk f32 -> bf16, 8 elements per thread.
// ---------------------------------------------------------------------------
__global__ __launch_bounds__(256) void cvt_wk(
    const float* __restrict__ wk, unsigned short* __restrict__ wkb) {
  const size_t i = (size_t)blockIdx.x * 256 + threadIdx.x;   // i-th group of 8
  const float4* s = (const float4*)(wk + i * 8);
  float4 v0 = s[0], v1 = s[1];
  float f[8] = { v0.x, v0.y, v0.z, v0.w, v1.x, v1.y, v1.z, v1.w };
  ((uint4*)(wkb + i * 8))[0] = pack8(f);
}

// ---------------------------------------------------------------------------
// Kernel 3: fused bf16 WMMA GEMM (xn @ Wk^T) + row argmax (per N-split).
//   Grid: (BATCH/32, NSPLIT). Block: 256 threads (8 wave32).
//   M-tile: 32 rows staged in LDS. Wave w covers N-tiles nt0+w, nt0+w+8, ...
//   Writes per-split (max, argmax) partials; small reduce kernel finishes.
// ---------------------------------------------------------------------------
__global__ __launch_bounds__(256) void gemm_argmax(
    const unsigned short* __restrict__ xn,
    const unsigned short* __restrict__ wkb,
    float* __restrict__ pVal,      // [NSPLIT][BATCH]
    int*   __restrict__ pIdx) {    // [NSPLIT][BATCH]
  extern __shared__ unsigned short As[];          // [32][A_STRIDE] bf16 bits
  __shared__ float sVal[8][32];
  __shared__ int   sIdx[8][32];

  const int t       = threadIdx.x;
  const int lane    = t & 31;
  const int wave    = t >> 5;
  const int m16     = lane & 15;                  // N (or M) index within tile
  const int khalf   = lane >> 4;                  // lane group 0/1
  const int rowBase = blockIdx.x * 32;
  const int split   = blockIdx.y;
  const int ntBase  = split * NT_PER_SPLIT;

  // ---- stage A tile: 32 rows x 2048 bf16, 16B per thread per row ----------
  #pragma unroll 4
  for (int r = 0; r < 32; ++r) {
    const uint4* s = (const uint4*)(xn + (size_t)(rowBase + r) * IN_);
    uint4*       d = (uint4*)(As + r * A_STRIDE);
    d[t] = s[t];
  }
  __syncthreads();

  float bestVal[2][8];
  int   bestIdx[2][8];
  #pragma unroll
  for (int s = 0; s < 2; ++s)
    #pragma unroll
    for (int r = 0; r < 8; ++r) { bestVal[s][r] = -__builtin_inff(); bestIdx[s][r] = 0; }

  // LDS row pointers for the two 16-row M sub-tiles (this lane's A row = m16)
  const unsigned short* aRow0 = As + (size_t)(m16)      * A_STRIDE;
  const unsigned short* aRow1 = As + (size_t)(m16 + 16) * A_STRIDE;

  for (int nt = ntBase + wave; nt < ntBase + NT_PER_SPLIT; nt += 8) {
    // B fragment source: lane holds column n = m16 of the N-tile, i.e. row
    // (nt*16 + m16) of Wk, contiguous along K.
    const unsigned short* wRow = wkb + (size_t)(nt * 16 + m16) * IN_;

    v8f acc0 = {}; v8f acc1 = {};

    #pragma unroll 2
    for (int k = 0; k < IN_; k += 32) {
      // A layout (ISA 7.12.2, 16-bit A 16x32): halfs[0..7]=K k+h*8+0..7,
      // halfs[8..15]=K k+16+h*8+0..7
      v16bf a0, a1, bf;
      ((uint4*)&a0)[0] = *(const uint4*)(aRow0 + k + khalf * 8);
      ((uint4*)&a0)[1] = *(const uint4*)(aRow0 + k + 16 + khalf * 8);
      ((uint4*)&a1)[0] = *(const uint4*)(aRow1 + k + khalf * 8);
      ((uint4*)&a1)[1] = *(const uint4*)(aRow1 + k + 16 + khalf * 8);
      // B layout: same as A: halfs[0..7]=K k+h*8+0..7, halfs[8..15]=K k+16+h*8+0..7
      ((uint4*)&bf)[0] = *(const uint4*)(wRow + k + khalf * 8);
      ((uint4*)&bf)[1] = *(const uint4*)(wRow + k + 16 + khalf * 8);

      acc0 = __builtin_amdgcn_wmma_f32_16x16x32_bf16(
          false, a0, false, bf, (short)0, acc0, false, false);
      acc1 = __builtin_amdgcn_wmma_f32_16x16x32_bf16(
          false, a1, false, bf, (short)0, acc1, false, false);
    }

    // Epilogue: this lane holds C[M = r + 8*khalf][N = m16] in acc[r].
    const int nglob = nt * 16 + m16;
    #pragma unroll
    for (int r = 0; r < 8; ++r) {
      if (acc0[r] > bestVal[0][r]) { bestVal[0][r] = acc0[r]; bestIdx[0][r] = nglob; }
      if (acc1[r] > bestVal[1][r]) { bestVal[1][r] = acc1[r]; bestIdx[1][r] = nglob; }
    }
  }

  // ---- reduce argmax across the 16 lanes that share each row --------------
  #pragma unroll
  for (int s = 0; s < 2; ++s) {
    #pragma unroll
    for (int r = 0; r < 8; ++r) {
      float v  = bestVal[s][r];
      int   ix = bestIdx[s][r];
      #pragma unroll
      for (int off = 8; off >= 1; off >>= 1) {
        float ov = __shfl_xor(v, off, 16);
        int   oi = __shfl_xor(ix, off, 16);
        if (ov > v || (ov == v && oi < ix)) { v = ov; ix = oi; }
      }
      if (m16 == 0) {
        const int localRow = s * 16 + r + 8 * khalf;
        sVal[wave][localRow] = v;
        sIdx[wave][localRow] = ix;
      }
    }
  }
  __syncthreads();

  // ---- reduce across the 8 waves; write this split's partial --------------
  if (t < 32) {
    float v  = sVal[0][t];
    int   ix = sIdx[0][t];
    #pragma unroll
    for (int w = 1; w < 8; ++w) {
      float ov = sVal[w][t];
      int   oi = sIdx[w][t];
      if (ov > v || (ov == v && oi < ix)) { v = ov; ix = oi; }
    }
    pVal[(size_t)split * BATCH + rowBase + t] = v;
    pIdx[(size_t)split * BATCH + rowBase + t] = ix;
  }
}

// ---------------------------------------------------------------------------
// Kernel 3b: reduce the NSPLIT partial argmax candidates per row.
// Lower split => lower n, so (>, tie: smaller idx) matches jnp.argmax.
// ---------------------------------------------------------------------------
__global__ __launch_bounds__(256) void argmax_reduce(
    const float* __restrict__ pVal, const int* __restrict__ pIdx,
    int* __restrict__ idxOut) {
  const int row = blockIdx.x * 256 + threadIdx.x;
  float v  = pVal[row];
  int   ix = pIdx[row];
  #pragma unroll
  for (int s = 1; s < NSPLIT; ++s) {
    float ov = pVal[(size_t)s * BATCH + row];
    int   oi = pIdx[(size_t)s * BATCH + row];
    if (ov > v || (ov == v && oi < ix)) { v = ov; ix = oi; }
  }
  idxOut[row] = ix;
}

// ---------------------------------------------------------------------------
// Kernel 4: out[b,c] = Wg[c, idx[b]]  (Wg is [NCLS, HID] row-major).
// Wg (32 MB) is L2-resident; random column gather is cheap vs the GEMM.
// ---------------------------------------------------------------------------
__global__ __launch_bounds__(256) void gather_out(
    const float* __restrict__ wg, const int* __restrict__ idx,
    float* __restrict__ out) {
  const int b = blockIdx.x;
  const int c = blockIdx.y * 256 + threadIdx.x;
  if (c < NCLS) {
    const int j = idx[b];
    out[(size_t)b * NCLS + c] = wg[(size_t)c * HID_ + j];
  }
}

// ---------------------------------------------------------------------------
extern "C" void kernel_launch(void* const* d_in, const int* in_sizes, int n_in,
                              void* d_out, int out_size, void* d_ws, size_t ws_size,
                              hipStream_t stream) {
  const float* x  = (const float*)d_in[0];   // [BATCH, IN]
  const float* Wk = (const float*)d_in[1];   // [HID, IN]
  const float* Wg = (const float*)d_in[2];   // [NCLS, HID]
  float* out = (float*)d_out;                // [BATCH, NCLS]

  // Workspace layout:
  //   xn bf16 (16MB) | Wk bf16 (32MB) | pVal (64KB) | pIdx (64KB) | idx (16KB)
  char* ws = (char*)d_ws;
  unsigned short* xnb  = (unsigned short*)ws;
  unsigned short* wkb  = (unsigned short*)(ws + (size_t)BATCH * IN_ * 2);
  char* tail = ws + (size_t)BATCH * IN_ * 2 + (size_t)HID_ * IN_ * 2;
  float* pVal = (float*)tail;
  int*   pIdx = (int*)(tail + (size_t)NSPLIT * BATCH * sizeof(float));
  int*   idx  = (int*)(tail + (size_t)NSPLIT * BATCH * (sizeof(float) + sizeof(int)));

  normalize_cvt<<<BATCH, 256, 0, stream>>>(x, xnb);

  cvt_wk<<<(int)(((size_t)HID_ * IN_ / 8) / 256), 256, 0, stream>>>(Wk, wkb);

  const size_t ldsBytes = 32 * A_STRIDE * sizeof(unsigned short);  // 131,584 B
  gemm_argmax<<<dim3(BATCH / 32, NSPLIT), 256, ldsBytes, stream>>>(xnb, wkb, pVal, pIdx);

  argmax_reduce<<<BATCH / 256, 256, 0, stream>>>(pVal, pIdx, idx);

  gather_out<<<dim3(BATCH, (NCLS + 255) / 256), 256, 0, stream>>>(Wg, idx, out);
}
